// GraphCastDecoder_86303072846452
// MI455X (gfx1250) — matrix operand order repaired and checked by reading, not compiled, not run on hardware
//
#include <hip/hip_runtime.h>
#include <math.h>

typedef __bf16 bf16;
typedef __attribute__((ext_vector_type(16))) __bf16 v16bf;
typedef __attribute__((ext_vector_type(8)))  __bf16 v8bf;
typedef __attribute__((ext_vector_type(8)))  float  v8f;

#define NGRID 65536
#define NEDGE 196608

__device__ __forceinline__ float silu_f(float x) {
  return x / (1.0f + __expf(-x));
}

__device__ __forceinline__ v16bf cvt16(v8f lo, v8f hi) {
  v16bf a;
#pragma unroll
  for (int i = 0; i < 8; ++i) {
    a[i]     = (bf16)lo[i];
    a[8 + i] = (bf16)hi[i];
  }
  return a;
}

// ---------------------------------------------------------------------------
// Wave GEMM, 32 rows x 128 cols, A in LDS (row-major bf16, 32 x K).
// acc[a*8+ct]: row fragment a (rows a*16..a*16+15), col tile ntBase+ct.
// B fragments pre-packed: frag f = kt*32+nt, 512 bf16, element = lane*16+i.
// Each B fragment load feeds TWO wmmas (row frags 0 and 1).
// ---------------------------------------------------------------------------
template <int K, int LDA>
__device__ __forceinline__ void wave_gemm2_lds(const bf16* __restrict__ sA,
                                               const bf16* __restrict__ Wp,
                                               int ntBase, int lane,
                                               v8f acc[16]) {
  const int row = lane & 15, half = lane >> 4;
  for (int kt = 0; kt < (K >> 5); ++kt) {
    const bf16* ap0 = sA + row * LDA + (kt << 5) + (half << 3);
    const bf16* ap1 = ap0 + 16 * LDA;
    v16bf a0, a1;
    ((v8bf*)&a0)[0] = *(const v8bf*)(ap0);
    ((v8bf*)&a0)[1] = *(const v8bf*)(ap0 + 16);
    ((v8bf*)&a1)[0] = *(const v8bf*)(ap1);
    ((v8bf*)&a1)[1] = *(const v8bf*)(ap1 + 16);
    const bf16* bp = Wp + ((size_t)(kt * 32 + ntBase) << 9) + (lane << 4);
#pragma unroll
    for (int ct = 0; ct < 8; ++ct) {
      const bf16* q = bp + (ct << 9);
      v16bf b;
      ((v8bf*)&b)[0] = *(const v8bf*)(q);
      ((v8bf*)&b)[1] = *(const v8bf*)(q + 8);
      acc[ct]     = __builtin_amdgcn_wmma_f32_16x16x32_bf16(
          false, a0, false, b, (short)0, acc[ct], false, false);
      acc[8 + ct] = __builtin_amdgcn_wmma_f32_16x16x32_bf16(
          false, a1, false, b, (short)0, acc[8 + ct], false, false);
    }
  }
}

// ---------------------------------------------------------------------------
// Same but A gathered directly from global fp32 rows (no LDS staging).
// pA0/pA1 are this lane's row-base pointers (rows row and row+16) into one
// 512-col source segment; nkt 32-wide K chunks starting at global kt=ktBase.
// ---------------------------------------------------------------------------
__device__ __forceinline__ void wave_gemm2_glob(const float* __restrict__ pA0,
                                                const float* __restrict__ pA1,
                                                const bf16* __restrict__ Wp,
                                                int ktBase, int nkt, int ntBase,
                                                int lane, v8f acc[16]) {
  const int half = lane >> 4;
  for (int k = 0; k < nkt; ++k) {
    const int off = (k << 5) + (half << 3);
    v8f lo0 = *(const v8f*)(pA0 + off);
    v8f hi0 = *(const v8f*)(pA0 + off + 16);
    v8f lo1 = *(const v8f*)(pA1 + off);
    v8f hi1 = *(const v8f*)(pA1 + off + 16);
    v16bf a0 = cvt16(lo0, hi0);
    v16bf a1 = cvt16(lo1, hi1);
    const bf16* bp =
        Wp + ((size_t)((ktBase + k) * 32 + ntBase) << 9) + (lane << 4);
#pragma unroll
    for (int ct = 0; ct < 8; ++ct) {
      const bf16* q = bp + (ct << 9);
      v16bf b;
      ((v8bf*)&b)[0] = *(const v8bf*)(q);
      ((v8bf*)&b)[1] = *(const v8bf*)(q + 8);
      acc[ct]     = __builtin_amdgcn_wmma_f32_16x16x32_bf16(
          false, a0, false, b, (short)0, acc[ct], false, false);
      acc[8 + ct] = __builtin_amdgcn_wmma_f32_16x16x32_bf16(
          false, a1, false, b, (short)0, acc[8 + ct], false, false);
    }
  }
}

// ---------------------------------------------------------------------------
// Block LayerNorm stats over 32 rows. Adds bias into acc; outputs per-row
// mean and rsqrt(var+eps). Lane owns rows a*16 + 8*(lane>>4) + r.
// sS/sQ: 4 waves x 32 rows.
// ---------------------------------------------------------------------------
__device__ __forceinline__ void block_ln_stats2(v8f acc[16],
                                                const float* __restrict__ b1,
                                                int colBase, int lane, int wv,
                                                float* sS, float* sQ,
                                                float mean[16], float inv[16]) {
  const int half = lane >> 4;
  float ps[16], pq[16];
#pragma unroll
  for (int j = 0; j < 16; ++j) { ps[j] = 0.f; pq[j] = 0.f; }
#pragma unroll
  for (int ct = 0; ct < 8; ++ct) {
    int col = colBase + (ct << 4) + (lane & 15);
    float bb = b1[col];
#pragma unroll
    for (int a = 0; a < 2; ++a)
#pragma unroll
      for (int r = 0; r < 8; ++r) {
        float v = acc[a * 8 + ct][r] + bb;
        acc[a * 8 + ct][r] = v;
        ps[a * 8 + r] += v;
        pq[a * 8 + r] += v * v;
      }
  }
  for (int m = 1; m <= 8; m <<= 1) {
#pragma unroll
    for (int j = 0; j < 16; ++j) {
      ps[j] += __shfl_xor(ps[j], m, 32);
      pq[j] += __shfl_xor(pq[j], m, 32);
    }
  }
  if ((lane & 15) == 0) {
#pragma unroll
    for (int a = 0; a < 2; ++a)
#pragma unroll
      for (int r = 0; r < 8; ++r) {
        int rowi = a * 16 + (half << 3) + r;
        sS[wv * 32 + rowi] = ps[a * 8 + r];
        sQ[wv * 32 + rowi] = pq[a * 8 + r];
      }
  }
  __syncthreads();
#pragma unroll
  for (int a = 0; a < 2; ++a)
#pragma unroll
    for (int r = 0; r < 8; ++r) {
      int rowi = a * 16 + (half << 3) + r;
      float s = sS[rowi] + sS[32 + rowi] + sS[64 + rowi] + sS[96 + rowi];
      float q = sQ[rowi] + sQ[32 + rowi] + sQ[64 + rowi] + sQ[96 + rowi];
      float mn = s * (1.0f / 512.0f);
      float var = q * (1.0f / 512.0f) - mn * mn;
      mean[a * 8 + r] = mn;
      inv[a * 8 + r] = rsqrtf(var + 1e-5f);
    }
}

// ---------------------------------------------------------------------------
// Pack fp32 weight [K,N] (row-major, N=512) into bf16 WMMA B fragments.
// Lane holds column n = nt*16 + (lane&15); element i -> k = kt*32+(i&7)+8*half+16*(i>>3).
// ---------------------------------------------------------------------------
__global__ void pack_weight_kernel(const float* __restrict__ W,
                                   bf16* __restrict__ out, int K, int N) {
  int idx = blockIdx.x * 256 + threadIdx.x;
  if (idx >= K * N) return;
  int f = idx >> 9, rem = idx & 511, lane = rem >> 4, i = rem & 15;
  int ntiles = N >> 4;
  int kt = f / ntiles, nt = f - kt * ntiles;
  int n = (nt << 4) + (lane & 15);
  int k = (kt << 5) + (i & 7) + ((lane >> 4) << 3) + ((i >> 3) << 4);
  out[idx] = (bf16)W[(size_t)k * N + n];
}

__global__ void zero_kernel(float* __restrict__ p, size_t n) {
  size_t i = (size_t)blockIdx.x * 256 + threadIdx.x;
  if (i < n) p[i] = 0.f;
}

// ---------------------------------------------------------------------------
// Embedder: [E,4] -> silu -> [E,512] -> LN. 32 edges per block, 128 threads.
// ---------------------------------------------------------------------------
__global__ void __launch_bounds__(128)
embed_kernel(const float* __restrict__ efeat, const float* __restrict__ W0,
             const float* __restrict__ b0, const bf16* __restrict__ W1p,
             const float* __restrict__ b1, const float* __restrict__ g,
             const float* __restrict__ bt, float* __restrict__ e_out) {
  __shared__ __align__(16) bf16 sH[32 * 512];
  __shared__ float sS[128];
  __shared__ float sQ[128];
  const int tid = threadIdx.x, lane = tid & 31, wv = tid >> 5;
  const int e0 = blockIdx.x << 5;
  for (int t = tid; t < 32 * 512; t += 128) {
    int r = t >> 9, c = t & 511;
    const float* x = efeat + (size_t)(e0 + r) * 4;
    float v = b0[c] + x[0] * W0[c] + x[1] * W0[512 + c] + x[2] * W0[1024 + c] +
              x[3] * W0[1536 + c];
    sH[t] = (bf16)silu_f(v);
  }
  __syncthreads();
  v8f zf = {0.f, 0.f, 0.f, 0.f, 0.f, 0.f, 0.f, 0.f};
  v8f acc[16];
#pragma unroll
  for (int j = 0; j < 16; ++j) acc[j] = zf;
  wave_gemm2_lds<512, 512>(sH, W1p, wv << 3, lane, acc);
  float mean[16], inv[16];
  block_ln_stats2(acc, b1, wv << 7, lane, wv, sS, sQ, mean, inv);
  const int half = lane >> 4;
  for (int ct = 0; ct < 8; ++ct) {
    int col = (wv << 7) + (ct << 4) + (lane & 15);
    float gg = g[col], be = bt[col];
#pragma unroll
    for (int a = 0; a < 2; ++a)
#pragma unroll
      for (int r = 0; r < 8; ++r) {
        int row = a * 16 + (half << 3) + r;
        e_out[(size_t)(e0 + row) * 512 + col] =
            (acc[a * 8 + ct][r] - mean[a * 8 + r]) * inv[a * 8 + r] * gg + be;
      }
  }
}

// ---------------------------------------------------------------------------
// Edge update: concat(e, mesh[src], grid[dst]) -> silu(.@W0+b0) -> .@W1+b1
// -> LN -> +e, atomic scatter-add into agg[dst]. 32 edges/block, no LDS
// staging for the concat input: stage-1 A fragments gathered from global.
// ---------------------------------------------------------------------------
__global__ void __launch_bounds__(128)
edge_kernel(const float* __restrict__ e_in, const float* __restrict__ mesh,
            const float* __restrict__ grid, const int* __restrict__ eidx,
            const bf16* __restrict__ W0p, const float* __restrict__ b0,
            const bf16* __restrict__ W1p, const float* __restrict__ b1,
            const float* __restrict__ g, const float* __restrict__ bt,
            float* __restrict__ agg) {
  __shared__ __align__(16) bf16 sH[32 * 512];
  __shared__ float sS[128];
  __shared__ float sQ[128];
  const int tid = threadIdx.x, lane = tid & 31, wv = tid >> 5;
  const int half = lane >> 4;
  const int e0 = blockIdx.x << 5;

  // per-lane row base pointers for its two row fragments
  const int eg0 = e0 + (lane & 15);
  const int eg1 = eg0 + 16;
  const float* pe0 = e_in + (size_t)eg0 * 512;
  const float* pe1 = e_in + (size_t)eg1 * 512;
  const float* pm0 = mesh + (size_t)eidx[eg0] * 512;
  const float* pm1 = mesh + (size_t)eidx[eg1] * 512;
  const float* pg0 = grid + (size_t)eidx[NEDGE + eg0] * 512;
  const float* pg1 = grid + (size_t)eidx[NEDGE + eg1] * 512;

  v8f zf = {0.f, 0.f, 0.f, 0.f, 0.f, 0.f, 0.f, 0.f};
  v8f acc[16];
#pragma unroll
  for (int j = 0; j < 16; ++j) acc[j] = zf;
  // stage 1: K = 1536 in three 512-wide segments
  wave_gemm2_glob(pe0, pe1, W0p, 0,  16, wv << 3, lane, acc);
  wave_gemm2_glob(pm0, pm1, W0p, 16, 16, wv << 3, lane, acc);
  wave_gemm2_glob(pg0, pg1, W0p, 32, 16, wv << 3, lane, acc);

  for (int ct = 0; ct < 8; ++ct) {
    int col = (wv << 7) + (ct << 4) + (lane & 15);
    float bb = b0[col];
#pragma unroll
    for (int a = 0; a < 2; ++a)
#pragma unroll
      for (int r = 0; r < 8; ++r)
        sH[(a * 16 + (half << 3) + r) * 512 + col] =
            (bf16)silu_f(acc[a * 8 + ct][r] + bb);
  }
  __syncthreads();
#pragma unroll
  for (int j = 0; j < 16; ++j) acc[j] = zf;
  wave_gemm2_lds<512, 512>(sH, W1p, wv << 3, lane, acc);
  float mean[16], inv[16];
  block_ln_stats2(acc, b1, wv << 7, lane, wv, sS, sQ, mean, inv);
  for (int ct = 0; ct < 8; ++ct) {
    int col = (wv << 7) + (ct << 4) + (lane & 15);
    float gg = g[col], be = bt[col];
#pragma unroll
    for (int a = 0; a < 2; ++a)
#pragma unroll
      for (int r = 0; r < 8; ++r) {
        int row = a * 16 + (half << 3) + r;
        int eg = e0 + row;
        float yn =
            (acc[a * 8 + ct][r] - mean[a * 8 + r]) * inv[a * 8 + r] * gg + be;
        float res = yn + e_in[(size_t)eg * 512 + col];
        unsafeAtomicAdd(&agg[(size_t)eidx[NEDGE + eg] * 512 + col], res);
      }
  }
}

// ---------------------------------------------------------------------------
// Node update: concat(grid, agg) -> silu(.@W0+b0) -> .@W1+b1 -> LN -> +grid.
// 32 nodes/block; stage-1 A fragments read directly from global.
// ---------------------------------------------------------------------------
__global__ void __launch_bounds__(128)
node_kernel(const float* __restrict__ grid, const float* __restrict__ agg,
            const bf16* __restrict__ W0p, const float* __restrict__ b0,
            const bf16* __restrict__ W1p, const float* __restrict__ b1,
            const float* __restrict__ g, const float* __restrict__ bt,
            float* __restrict__ out) {
  __shared__ __align__(16) bf16 sH[32 * 512];
  __shared__ float sS[128];
  __shared__ float sQ[128];
  const int tid = threadIdx.x, lane = tid & 31, wv = tid >> 5;
  const int half = lane >> 4;
  const int n0 = blockIdx.x << 5;

  const int ng0 = n0 + (lane & 15);
  const int ng1 = ng0 + 16;
  const float* pG0 = grid + (size_t)ng0 * 512;
  const float* pG1 = grid + (size_t)ng1 * 512;
  const float* pA0 = agg + (size_t)ng0 * 512;
  const float* pA1 = agg + (size_t)ng1 * 512;

  v8f zf = {0.f, 0.f, 0.f, 0.f, 0.f, 0.f, 0.f, 0.f};
  v8f acc[16];
#pragma unroll
  for (int j = 0; j < 16; ++j) acc[j] = zf;
  // stage 1: K = 1024 in two 512-wide segments
  wave_gemm2_glob(pG0, pG1, W0p, 0,  16, wv << 3, lane, acc);
  wave_gemm2_glob(pA0, pA1, W0p, 16, 16, wv << 3, lane, acc);

  for (int ct = 0; ct < 8; ++ct) {
    int col = (wv << 7) + (ct << 4) + (lane & 15);
    float bb = b0[col];
#pragma unroll
    for (int a = 0; a < 2; ++a)
#pragma unroll
      for (int r = 0; r < 8; ++r)
        sH[(a * 16 + (half << 3) + r) * 512 + col] =
            (bf16)silu_f(acc[a * 8 + ct][r] + bb);
  }
  __syncthreads();
#pragma unroll
  for (int j = 0; j < 16; ++j) acc[j] = zf;
  wave_gemm2_lds<512, 512>(sH, W1p, wv << 3, lane, acc);
  float mean[16], inv[16];
  block_ln_stats2(acc, b1, wv << 7, lane, wv, sS, sQ, mean, inv);
  for (int ct = 0; ct < 8; ++ct) {
    int col = (wv << 7) + (ct << 4) + (lane & 15);
    float gg = g[col], be = bt[col];
#pragma unroll
    for (int a = 0; a < 2; ++a)
#pragma unroll
      for (int r = 0; r < 8; ++r) {
        int row = a * 16 + (half << 3) + r;
        float yn =
            (acc[a * 8 + ct][r] - mean[a * 8 + r]) * inv[a * 8 + r] * gg + be;
        out[(size_t)(n0 + row) * 512 + col] =
            yn + grid[(size_t)(n0 + row) * 512 + col];
      }
  }
}

extern "C" void kernel_launch(void* const* d_in, const int* in_sizes, int n_in,
                              void* d_out, int out_size, void* d_ws,
                              size_t ws_size, hipStream_t stream) {
  (void)in_sizes; (void)n_in; (void)out_size; (void)ws_size;
  const float* grid_nfeat = (const float*)d_in[0];
  const float* mesh_nfeat = (const float*)d_in[1];
  const int*   edge_index = (const int*)d_in[2];
  const float* efeat      = (const float*)d_in[3];
  const float* emb_W0 = (const float*)d_in[4];
  const float* emb_b0 = (const float*)d_in[5];
  const float* emb_W1 = (const float*)d_in[6];
  const float* emb_b1 = (const float*)d_in[7];
  const float* emb_g  = (const float*)d_in[8];
  const float* emb_b  = (const float*)d_in[9];
  const float* edge_W0 = (const float*)d_in[10];
  const float* edge_b0 = (const float*)d_in[11];
  const float* edge_W1 = (const float*)d_in[12];
  const float* edge_b1 = (const float*)d_in[13];
  const float* edge_g  = (const float*)d_in[14];
  const float* edge_beta = (const float*)d_in[15];
  const float* node_W0 = (const float*)d_in[16];
  const float* node_b0 = (const float*)d_in[17];
  const float* node_W1 = (const float*)d_in[18];
  const float* node_b1 = (const float*)d_in[19];
  const float* node_g  = (const float*)d_in[20];
  const float* node_beta = (const float*)d_in[21];

  // Workspace layout
  float* e   = (float*)d_ws;                       // [E,512] fp32
  float* agg = e + (size_t)NEDGE * 512;            // [NG,512] fp32
  bf16* embW1p  = (bf16*)(agg + (size_t)NGRID * 512);
  bf16* edgeW0p = embW1p + 512 * 512;
  bf16* edgeW1p = edgeW0p + 1536 * 512;
  bf16* nodeW0p = edgeW1p + 512 * 512;
  bf16* nodeW1p = nodeW0p + 1024 * 512;

  pack_weight_kernel<<<(512 * 512 + 255) / 256, 256, 0, stream>>>(emb_W1, embW1p, 512, 512);
  pack_weight_kernel<<<(1536 * 512 + 255) / 256, 256, 0, stream>>>(edge_W0, edgeW0p, 1536, 512);
  pack_weight_kernel<<<(512 * 512 + 255) / 256, 256, 0, stream>>>(edge_W1, edgeW1p, 512, 512);
  pack_weight_kernel<<<(1024 * 512 + 255) / 256, 256, 0, stream>>>(node_W0, nodeW0p, 1024, 512);
  pack_weight_kernel<<<(512 * 512 + 255) / 256, 256, 0, stream>>>(node_W1, nodeW1p, 512, 512);

  size_t aggN = (size_t)NGRID * 512;
  zero_kernel<<<(unsigned)((aggN + 255) / 256), 256, 0, stream>>>(agg, aggN);

  embed_kernel<<<NEDGE / 32, 128, 0, stream>>>(efeat, emb_W0, emb_b0, embW1p,
                                               emb_b1, emb_g, emb_b, e);
  edge_kernel<<<NEDGE / 32, 128, 0, stream>>>(e, mesh_nfeat, grid_nfeat,
                                              edge_index, edgeW0p, edge_b0,
                                              edgeW1p, edge_b1, edge_g,
                                              edge_beta, agg);
  node_kernel<<<NGRID / 32, 128, 0, stream>>>(grid_nfeat, agg, nodeW0p,
                                              node_b0, nodeW1p, node_b1,
                                              node_g, node_beta,
                                              (float*)d_out);
}
